// FoSae_31241592111346
// MI455X (gfx1250) — matrix operand order repaired and checked by reading, not compiled, not run on hardware
//
#include <hip/hip_runtime.h>
#include <hip/hip_bf16.h>

typedef __attribute__((ext_vector_type(16))) __bf16 v16bf;
typedef __attribute__((ext_vector_type(8)))  __bf16 v8bf;
typedef __attribute__((ext_vector_type(8)))  float  v8f;

// Problem constants
#define BB 16
#define NN 8
#define CC 3
#define HH 64
#define WW 96
#define POS 384          // 16*24 conv output positions
#define K_BIN 384        // 6*8*8
#define K_UN 192         // 3*8*8
#define H1DIM 6144       // 16*384
#define FCP 112          // FC=100 padded to 7*16
#define NSAMP_B 3072     // 3*16*64
#define NSAMP_U 384      // 3*16*8

// LDS tile geometry: 36 padded rows x 104 padded cols per channel (bf16)
#define LROWS 36
#define LPITCH 104
#define LCH (LROWS * LPITCH)   // 3744 elements per channel

// ---------------- workspace layout (bytes) ----------------
static constexpr size_t OFF_ACONV_B = 0;         // 12*32*16 bf16      (12288 B)
static constexpr size_t OFF_ACONV_U = 12288;     // 6*32*16 bf16       (6144 B)
static constexpr size_t OFF_W2P_B   = 18432;     // 7*192*512 bf16     (1376256 B)
static constexpr size_t OFF_W2P_U   = 1394688;   // 7*192*512 bf16     (1376256 B)
static constexpr size_t OFF_H1_B    = 2770944;   // 3072*6144 bf16     (37748736 B)
static constexpr size_t OFF_H1_U    = 40519680;  // 384*6144 bf16      (4718592 B)
static constexpr size_t OFF_H2_B    = 45238272;  // 3072*112 bf16      (688128 B)
static constexpr size_t OFF_H2_U    = 45926400;  // 384*112 bf16       (86016 B)
static constexpr size_t OFF_PB      = 46012416;  // 3072*2 f32         (24576 B)
static constexpr size_t OFF_PU      = 46036992;  // 384*2 f32          (3072 B)

// ---------------------------------------------------------------------------
// Pack conv weights [OC=16][K] (f32) into WMMA A-fragment order (bf16).
// A 16x32 bf16 k order used here: lane<16 -> M=lane, K = base+0..15
//                      lane>=16 -> M=lane-16, K = base+16..31
__global__ void pack_conv_a(const float* __restrict__ Wc, __bf16* __restrict__ dst,
                            int nChunks, int K) {
  int t = blockIdx.x * blockDim.x + threadIdx.x;
  int total = nChunks * 32 * 16;
  if (t >= total) return;
  int j = t & 15;
  int l = (t >> 4) & 31;
  int c = t >> 9;
  bool hi = l >= 16;
  int k = c * 32 + (hi ? 16 : 0) + j;
  int oc = l & 15;
  dst[t] = (__bf16)Wc[oc * K + k];
}

// Pack FC weights W[K][Ncols] (f32) into WMMA B-fragment order (bf16), padded.
// B 32x16 bf16 layout: lane<16 -> col=lane, K=base+{0..7,16..23}
//                      lane>=16 -> col=lane-16, K=base+{8..15,24..31}
__global__ void pack_fc_b(const float* __restrict__ W, __bf16* __restrict__ dst,
                          int ntiles, int kchunks, int ncols, int krows) {
  int t = blockIdx.x * blockDim.x + threadIdx.x;
  int total = ntiles * kchunks * 512;
  if (t >= total) return;
  int j = t & 15;
  int l = (t >> 4) & 31;
  int kc = (t >> 9) % kchunks;
  int nt = t / (kchunks * 512);
  int col = nt * 16 + (l & 15);
  int k = kc * 32 + (l >= 16 ? 8 : 0) + (j < 8 ? j : j + 8);
  float v = 0.0f;
  if (col < ncols && k < krows) v = W[(size_t)k * ncols + col];
  dst[t] = (__bf16)v;
}

// ---------------------------------------------------------------------------
// Conv layer-1 as implicit GEMM with LDS-staged zero-padded bf16 image.
// One workgroup (4 waves) = one (state, sample, half of output rows).
// A fragments (weights) preloaded to registers once per wave; B fragments are
// branch-free ds_load_b64 pairs from the padded LDS tile.

__global__ void __launch_bounds__(128) conv_bin(
    const float* __restrict__ s0, const float* __restrict__ s1,
    const float* __restrict__ s2, const __bf16* __restrict__ Apack,
    const float* __restrict__ bc, __bf16* __restrict__ h1) {
  __shared__ __align__(16) __bf16 sm[6 * LCH];   // 44928 B

  int w = blockIdx.x;
  int half = w & 1;
  int sample = (w >> 1) & 1023;
  int st = w >> 11;
  const float* S = (st == 0) ? s0 : ((st == 1) ? s1 : s2);
  int b = sample >> 6, pair = sample & 63;
  int iobj = pair >> 3, jobj = pair & 7;

  int lane = threadIdx.x & 31;

  // ---- preload all 12 A fragments (weights) into registers ----
  v16bf afrag[12];
#pragma unroll
  for (int c = 0; c < 12; ++c)
    afrag[c] = *(const v16bf*)(Apack + (size_t)(c * 32 + lane) * 16);

  // ---- fill padded image tile (zeros in halo) ----
  int iybase = 32 * half - 2;
  for (int idx = threadIdx.x; idx < 6 * LCH; idx += 128) {
    int ic = idx / LCH;
    int rem = idx - ic * LCH;
    int ly = rem / LPITCH, lx = rem - ly * LPITCH;
    int iy = iybase + ly;
    int ix = lx - 2;
    float v = 0.0f;
    if ((unsigned)iy < (unsigned)HH && (unsigned)ix < (unsigned)WW) {
      int obj = (ic < 3) ? iobj : jobj;
      int ch  = (ic < 3) ? ic : ic - 3;
      v = S[((((size_t)b * NN + obj) * CC + ch) * HH + iy) * WW + ix];
    }
    sm[idx] = (__bf16)v;
  }
  __syncthreads();

  // ---- GEMM: M=16 oc, N=16 positions, K=384 ----
  int wv = threadIdx.x >> 5;
  bool hi = lane >= 16;
  int col = lane & 15;
  size_t srow = (size_t)(st * 1024 + sample) * H1DIM;

  for (int t = 0; t < 3; ++t) {
    int ptl = wv * 3 + t;                 // 0..11 within this half
    int p = half * 192 + ptl * 16 + col;  // global position
    int py = p / 24, px = p % 24;
    int ly0 = 4 * (py - 8 * half);        // 0..28
    int lx0 = px * 4;                     // 8-byte aligned in bf16

    v8f acc;
#pragma unroll
    for (int i = 0; i < 8; ++i) acc[i] = 0.0f;

#pragma unroll
    for (int c = 0; c < 12; ++c) {
      int kbase = c * 32 + (hi ? 16 : 0);
      int ic = kbase >> 6;
      int ky0 = (kbase & 63) >> 3;
      const __bf16* r0 = sm + ic * LCH + (ly0 + ky0) * LPITCH + lx0;
      union { v16bf v; uint2 q[4]; } bu;
      bu.q[0] = ((const uint2*)r0)[0];
      bu.q[1] = ((const uint2*)r0)[1];
      bu.q[2] = ((const uint2*)(r0 + LPITCH))[0];
      bu.q[3] = ((const uint2*)(r0 + LPITCH))[1];
      acc = __builtin_amdgcn_wmma_f32_16x16x32_bf16(false, afrag[c], false, bu.v,
                                                    (short)0, acc, false, false);
    }

    int ocb = hi ? 8 : 0;
#pragma unroll
    for (int r = 0; r < 8; ++r) {
      int oc = ocb + r;
      float v = acc[r] + bc[oc];
      v = v > 0.0f ? v : 0.0f;
      h1[srow + (size_t)oc * POS + p] = (__bf16)v;
    }
  }
}

__global__ void __launch_bounds__(128) conv_un(
    const float* __restrict__ s0, const float* __restrict__ s1,
    const float* __restrict__ s2, const __bf16* __restrict__ Apack,
    const float* __restrict__ bc, __bf16* __restrict__ h1) {
  __shared__ __align__(16) __bf16 sm[3 * LCH];   // 22464 B

  int w = blockIdx.x;
  int half = w & 1;
  int sample = (w >> 1) & 127;
  int st = (w >> 1) >> 7;
  const float* S = (st == 0) ? s0 : ((st == 1) ? s1 : s2);
  int b = sample >> 3, obj = sample & 7;

  int lane = threadIdx.x & 31;

  v16bf afrag[6];
#pragma unroll
  for (int c = 0; c < 6; ++c)
    afrag[c] = *(const v16bf*)(Apack + (size_t)(c * 32 + lane) * 16);

  int iybase = 32 * half - 2;
  for (int idx = threadIdx.x; idx < 3 * LCH; idx += 128) {
    int ic = idx / LCH;
    int rem = idx - ic * LCH;
    int ly = rem / LPITCH, lx = rem - ly * LPITCH;
    int iy = iybase + ly;
    int ix = lx - 2;
    float v = 0.0f;
    if ((unsigned)iy < (unsigned)HH && (unsigned)ix < (unsigned)WW)
      v = S[((((size_t)b * NN + obj) * CC + ic) * HH + iy) * WW + ix];
    sm[idx] = (__bf16)v;
  }
  __syncthreads();

  int wv = threadIdx.x >> 5;
  bool hi = lane >= 16;
  int col = lane & 15;
  size_t srow = (size_t)(st * 128 + sample) * H1DIM;

  for (int t = 0; t < 3; ++t) {
    int ptl = wv * 3 + t;
    int p = half * 192 + ptl * 16 + col;
    int py = p / 24, px = p % 24;
    int ly0 = 4 * (py - 8 * half);
    int lx0 = px * 4;

    v8f acc;
#pragma unroll
    for (int i = 0; i < 8; ++i) acc[i] = 0.0f;

#pragma unroll
    for (int c = 0; c < 6; ++c) {
      int kbase = c * 32 + (hi ? 16 : 0);
      int ic = kbase >> 6;
      int ky0 = (kbase & 63) >> 3;
      const __bf16* r0 = sm + ic * LCH + (ly0 + ky0) * LPITCH + lx0;
      union { v16bf v; uint2 q[4]; } bu;
      bu.q[0] = ((const uint2*)r0)[0];
      bu.q[1] = ((const uint2*)r0)[1];
      bu.q[2] = ((const uint2*)(r0 + LPITCH))[0];
      bu.q[3] = ((const uint2*)(r0 + LPITCH))[1];
      acc = __builtin_amdgcn_wmma_f32_16x16x32_bf16(false, afrag[c], false, bu.v,
                                                    (short)0, acc, false, false);
    }

    int ocb = hi ? 8 : 0;
#pragma unroll
    for (int r = 0; r < 8; ++r) {
      int oc = ocb + r;
      float v = acc[r] + bc[oc];
      v = v > 0.0f ? v : 0.0f;
      h1[srow + (size_t)oc * POS + p] = (__bf16)v;
    }
  }
}

// ---------------------------------------------------------------------------
// FC1: h2 = relu(h1 @ W2 + b2). [samples x 6144] @ [6144 x 112(pad)].
__global__ void fc1(const __bf16* __restrict__ h1, const __bf16* __restrict__ W2p,
                    const float* __restrict__ b2, __bf16* __restrict__ h2, int mtiles) {
  int wave = blockIdx.x * (blockDim.x >> 5) + (threadIdx.x >> 5);
  int lane = threadIdx.x & 31;
  int ntile = wave % 7;
  int mtile = wave / 7;
  if (mtile >= mtiles) return;
  bool hi = lane >= 16;
  const __bf16* arow = h1 + (size_t)(mtile * 16 + (lane & 15)) * H1DIM;

  v8f acc;
#pragma unroll
  for (int i = 0; i < 8; ++i) acc[i] = 0.0f;

  for (int kc = 0; kc < 192; ++kc) {
    int o0 = kc * 32 + (hi ? 8 : 0);
    __builtin_prefetch(arow + o0 + 64, 0, 0);   // gfx1250 global_prefetch
    v8bf a0 = *(const v8bf*)(arow + o0);
    v8bf a1 = *(const v8bf*)(arow + o0 + 16);
    v16bf a = __builtin_shufflevector(a0, a1, 0, 1, 2, 3, 4, 5, 6, 7,
                                      8, 9, 10, 11, 12, 13, 14, 15);
    v16bf bf = *(const v16bf*)(W2p + ((size_t)(ntile * 192 + kc) * 32 + lane) * 16);
    acc = __builtin_amdgcn_wmma_f32_16x16x32_bf16(false, a, false, bf,
                                                  (short)0, acc, false, false);
  }

  int col = ntile * 16 + (lane & 15);
  float bias = (col < 100) ? b2[col] : 0.0f;
  int rb = hi ? 8 : 0;
#pragma unroll
  for (int r = 0; r < 8; ++r) {
    int s = mtile * 16 + rb + r;
    float v = acc[r] + bias;
    v = v > 0.0f ? v : 0.0f;
    h2[(size_t)s * FCP + col] = (__bf16)v;
  }
}

// ---------------------------------------------------------------------------
// FC2 (100 -> 2) + gumbel-softmax. 200 MACs/sample: VALU path.
__global__ void fc2_softmax(const __bf16* __restrict__ h2, const float* __restrict__ W3,
                            const float* __restrict__ b3, const float* __restrict__ g,
                            const float* __restrict__ temp, float* __restrict__ probs,
                            int nSamples) {
  int t = blockIdx.x * blockDim.x + threadIdx.x;
  if (t >= nSamples) return;
  const __bf16* row = h2 + (size_t)t * FCP;
  float l0 = b3[0], l1 = b3[1];
#pragma unroll 4
  for (int k = 0; k < 100; ++k) {
    float h = (float)row[k];
    l0 += h * W3[k * 2];
    l1 += h * W3[k * 2 + 1];
  }
  float it = 1.0f / temp[0];
  float a0 = (l0 + g[t * 2]) * it;
  float a1 = (l1 + g[t * 2 + 1]) * it;
  float m = fmaxf(a0, a1);
  float e0 = __expf(a0 - m), e1 = __expf(a1 - m);
  float inv = 1.0f / (e0 + e1);
  probs[t * 2]     = e0 * inv;
  probs[t * 2 + 1] = e1 * inv;
}

// ---------------------------------------------------------------------------
// Final compact/mask gather into d_out.
__global__ void assemble(const float* __restrict__ pb, const float* __restrict__ pu,
                         const int* __restrict__ nobj, float* __restrict__ out) {
  int t = blockIdx.x * blockDim.x + threadIdx.x;
  if (t < 3072) {                       // binary: 3 states * 16 * 64
    int st = t / 1024;
    int r = t % 1024;
    int b = r >> 6, p = r & 63;
    int n = nobj[b];
    int ns = n < 1 ? 1 : n;
    int k = (p / ns) * NN + (p % ns);
    if (k > 63) k = 63;
    bool valid = p < n * n;
    float v0 = 0.0f, v1 = 0.0f;
    if (valid) {
      const float* s = pb + (size_t)(st * 1024 + b * 64 + k) * 2;
      v0 = s[0]; v1 = s[1];
    }
    float* o = out + (size_t)st * 2048 + (size_t)(b * 64 + p) * 2;
    o[0] = v0; o[1] = v1;
  } else if (t < 3072 + 384) {          // unary: 3 states * 16 * 8
    int u = t - 3072;
    int st = u / 128;
    int r = u % 128;
    int b = r >> 3, j = r & 7;
    int n = nobj[b];
    float v0 = 0.0f, v1 = 0.0f;
    if (j < n) {
      const float* s = pu + (size_t)(st * 128 + b * 8 + j) * 2;
      v0 = s[0]; v1 = s[1];
    }
    float* o = out + 6144 + (size_t)st * 256 + (size_t)(b * 8 + j) * 2;
    o[0] = v0; o[1] = v1;
  }
}

// ---------------------------------------------------------------------------
extern "C" void kernel_launch(void* const* d_in, const int* in_sizes, int n_in,
                              void* d_out, int out_size, void* d_ws, size_t ws_size,
                              hipStream_t stream) {
  const float* state       = (const float*)d_in[0];
  const float* state_next  = (const float*)d_in[1];
  const float* state_tilda = (const float*)d_in[2];
  const int*   n_obj       = (const int*)d_in[3];
  const float* temp        = (const float*)d_in[4];
  const float* g_unary     = (const float*)d_in[5];
  const float* g_binary    = (const float*)d_in[6];
  const float* Wc_u = (const float*)d_in[7];
  const float* bc_u = (const float*)d_in[8];
  const float* W2_u = (const float*)d_in[9];
  const float* b2_u = (const float*)d_in[10];
  const float* W3_u = (const float*)d_in[11];
  const float* b3_u = (const float*)d_in[12];
  const float* Wc_b = (const float*)d_in[13];
  const float* bc_b = (const float*)d_in[14];
  const float* W2_b = (const float*)d_in[15];
  const float* b2_b = (const float*)d_in[16];
  const float* W3_b = (const float*)d_in[17];
  const float* b3_b = (const float*)d_in[18];

  char* ws = (char*)d_ws;
  __bf16* Aconv_b = (__bf16*)(ws + OFF_ACONV_B);
  __bf16* Aconv_u = (__bf16*)(ws + OFF_ACONV_U);
  __bf16* W2p_b   = (__bf16*)(ws + OFF_W2P_B);
  __bf16* W2p_u   = (__bf16*)(ws + OFF_W2P_U);
  __bf16* h1_b    = (__bf16*)(ws + OFF_H1_B);
  __bf16* h1_u    = (__bf16*)(ws + OFF_H1_U);
  __bf16* h2_b    = (__bf16*)(ws + OFF_H2_B);
  __bf16* h2_u    = (__bf16*)(ws + OFF_H2_U);
  float*  probs_b = (float*)(ws + OFF_PB);
  float*  probs_u = (float*)(ws + OFF_PU);
  float*  out     = (float*)d_out;

  // Weight packing (bf16, WMMA fragment order)
  pack_conv_a<<<24, 256, 0, stream>>>(Wc_b, Aconv_b, 12, K_BIN);
  pack_conv_a<<<12, 256, 0, stream>>>(Wc_u, Aconv_u, 6, K_UN);
  pack_fc_b<<<2688, 256, 0, stream>>>(W2_b, W2p_b, 7, 192, 100, H1DIM);
  pack_fc_b<<<2688, 256, 0, stream>>>(W2_u, W2p_u, 7, 192, 100, H1DIM);

  // Conv layer: LDS-staged implicit GEMM (v_wmma bf16 + ds_load_b64 fragments)
  conv_bin<<<6144, 128, 0, stream>>>(state, state_next, state_tilda, Aconv_b, bc_b, h1_b);
  conv_un<<<768, 128, 0, stream>>>(state, state_next, state_tilda, Aconv_u, bc_u, h1_u);

  // FC1 (WMMA GEMM): binary 1344 waves, unary 168 waves
  fc1<<<168, 256, 0, stream>>>(h1_b, W2p_b, b2_b, h2_b, 192);
  fc1<<<21, 256, 0, stream>>>(h1_u, W2p_u, b2_u, h2_u, 24);

  // FC2 + gumbel softmax
  fc2_softmax<<<12, 256, 0, stream>>>(h2_b, W3_b, b3_b, g_binary, temp, probs_b, NSAMP_B);
  fc2_softmax<<<2, 256, 0, stream>>>(h2_u, W3_u, b3_u, g_unary, temp, probs_u, NSAMP_U);

  // Compact + validity masks into d_out
  assemble<<<14, 256, 0, stream>>>(probs_b, probs_u, n_obj, out);
}
